// SimplifiedPointNet_46076409152323
// MI455X (gfx1250) — compile-verified, matrix operand order checked by reading, not run on hardware
//
#include <hip/hip_runtime.h>
#include <hip/hip_bf16.h>

typedef __attribute__((ext_vector_type(16))) __bf16 v16bf;
typedef __attribute__((ext_vector_type(8)))  float  v8f;

#define BB   32
#define NN   32768
#define GG   8
#define GV   512
#define EPS  1e-5f

// ---------------------------------------------------------------------------
// zero init for gf (voxel features) + acc (fc1 accumulator)
// ---------------------------------------------------------------------------
__global__ void zero_kernel(float* __restrict__ p, int n) {
    int i = blockIdx.x * blockDim.x + threadIdx.x;
    if (i < n) p[i] = 0.0f;
}

// ---------------------------------------------------------------------------
// Kernel A: per-point MLP (3->64 VALU, 64->128 WMMA bf16) + voxel atomic-max.
// 256 threads = 8 waves; 256 points per block; 16 chunks of 16 points, two
// chunks per wave interleaved inside the N-tile loop so each B fragment is
// loaded once and the 4 WMMAs per tile cover each other's result hazards.
//
// Fragment addressing (ISA 7.12.2, 16-bit A 16x32):
//   lane = kh*16 + m   (kh = (kk>>3)&1,  kk = k % 32)
//   slot = 8*(kk>=16) + (kk & 7)         (slot = element index in v16bf)
// ---------------------------------------------------------------------------
__global__ __launch_bounds__(256) void mlp_pool_kernel(
    const float* __restrict__ points,
    const float* __restrict__ w1, const float* __restrict__ b1,
    const float* __restrict__ g1, const float* __restrict__ be1,
    const float* __restrict__ m1, const float* __restrict__ v1,
    const float* __restrict__ w2, const float* __restrict__ b2,
    const float* __restrict__ g2, const float* __restrict__ be2,
    const float* __restrict__ m2, const float* __restrict__ v2,
    float* __restrict__ gf)               // (B*GV, 128) zero-initialized
{
    __shared__ __bf16 sh_hfrag[16][2][32][16]; // [chunk][khalf][lane][slot] 32KB
    __shared__ __bf16 sh_bfrag[2][8][32][16];  // [khalf][nt][lane][slot]    16KB
    __shared__ float  sh_A1[64], sh_B1[64];
    __shared__ float  sh_A2[128], sh_B2[128];
    __shared__ int    sh_gfbase[256];

    const int tid = threadIdx.x;

    // fold bias + batchnorm into scale/shift
    if (tid < 64) {
        float s = g1[tid] * rsqrtf(v1[tid] + EPS);
        sh_A1[tid] = s;
        sh_B1[tid] = (b1[tid] - m1[tid]) * s + be1[tid];
    }
    if (tid < 128) {
        float s = g2[tid] * rsqrtf(v2[tid] + EPS);
        sh_A2[tid] = s;
        sh_B2[tid] = (b2[tid] - m2[tid]) * s + be2[tid];
    }

    // w2 (64x128 f32) -> bf16 LDS in fragment order, packed pair stores
    for (int i = tid; i < 2 * 8 * 32 * 16 / 2; i += 256) {
        const int ii    = i * 2;
        const int slot  = ii & 15;            // even
        const int lane  = (ii >> 4) & 31;
        const int nt    = (ii >> 9) & 7;
        const int khalf = (ii >> 12) & 1;
        const int n  = nt * 16 + (lane & 15);
        const int kh = lane >> 4;
        const int k0 = khalf * 32 + ((slot & 8) ? 16 : 0) + kh * 8 + (slot & 6);
        union { __bf16 h[2]; unsigned u; } pk;
        pk.h[0] = (__bf16)w2[k0 * 128 + n];
        pk.h[1] = (__bf16)w2[(k0 + 1) * 128 + n];
        *(unsigned*)&sh_bfrag[khalf][nt][lane][slot] = pk.u;
    }
    __syncthreads();

    // ---- phase 1: per-lane layer 1 + voxel index ----
    const int p = blockIdx.x * 256 + tid;           // global point id
    const int b = p / NN;
    const float px = points[p * 3 + 0];
    const float py = points[p * 3 + 1];
    const float pz = points[p * 3 + 2];

    int gx = (int)floorf((px + 1.0f) * 0.5f * (GG - 1e-5f));
    int gy = (int)floorf((py + 1.0f) * 0.5f * (GG - 1e-5f));
    int gz = (int)floorf((pz + 1.0f) * 0.5f * (GG - 1e-5f));
    gx = min(max(gx, 0), GG - 1);
    gy = min(max(gy, 0), GG - 1);
    gz = min(max(gz, 0), GG - 1);
    sh_gfbase[tid] = (b * GV + (gx * 64 + gy * 8 + gz)) * 128;

    {
        const int mychunk = tid >> 4, mrow = tid & 15;
        __bf16* hb = &sh_hfrag[mychunk][0][0][0];
        #pragma unroll
        for (int c = 0; c < 64; c += 2) {
            float y0 = px * w1[c]     + py * w1[64 + c]     + pz * w1[128 + c];
            float y1 = px * w1[c + 1] + py * w1[64 + c + 1] + pz * w1[128 + c + 1];
            y0 = fmaxf(y0 * sh_A1[c]     + sh_B1[c],     0.0f);
            y1 = fmaxf(y1 * sh_A1[c + 1] + sh_B1[c + 1], 0.0f);
            const int kk    = c & 31;                 // even
            const int khalf = c >> 5;
            const int kh    = (kk >> 3) & 1;
            const int slot  = ((kk >> 4) & 1) * 8 + (kk & 7);
            union { __bf16 h[2]; unsigned u; } pk;
            pk.h[0] = (__bf16)y0; pk.h[1] = (__bf16)y1;
            *(unsigned*)&hb[(khalf * 32 + kh * 16 + mrow) * 16 + slot] = pk.u;
        }
    }
    __syncthreads();

    // ---- phase 2: 16x64 @ 64x128 via WMMA, then bn/relu + atomic max ----
    const int wave = tid >> 5;
    const int lane = tid & 31;
    const int mcol = lane & 15;
    const int kh   = lane >> 4;

    // hoist per-lane BN constants (column depends only on nt and lane)
    float csA[8], csB[8];
    #pragma unroll
    for (int nt = 0; nt < 8; ++nt) {
        csA[nt] = sh_A2[nt * 16 + mcol];
        csB[nt] = sh_B2[nt * 16 + mcol];
    }

    const int chunk0 = wave * 2;
    const int chunk1 = chunk0 + 1;

    const v16bf a00 = *(const v16bf*)&sh_hfrag[chunk0][0][lane][0];
    const v16bf a01 = *(const v16bf*)&sh_hfrag[chunk0][1][lane][0];
    const v16bf a10 = *(const v16bf*)&sh_hfrag[chunk1][0][lane][0];
    const v16bf a11 = *(const v16bf*)&sh_hfrag[chunk1][1][lane][0];

    int gbase0[8], gbase1[8];
    #pragma unroll
    for (int r = 0; r < 8; ++r) {
        gbase0[r] = sh_gfbase[chunk0 * 16 + r + 8 * kh];
        gbase1[r] = sh_gfbase[chunk1 * 16 + r + 8 * kh];
    }

    #pragma unroll
    for (int nt = 0; nt < 8; ++nt) {
        const v16bf bf0 = *(const v16bf*)&sh_bfrag[0][nt][lane][0];
        const v16bf bf1 = *(const v16bf*)&sh_bfrag[1][nt][lane][0];

        v8f c0 = {}, c1 = {};
        c0 = __builtin_amdgcn_wmma_f32_16x16x32_bf16(false, a00, false, bf0, (short)0, c0, false, false);
        c1 = __builtin_amdgcn_wmma_f32_16x16x32_bf16(false, a10, false, bf0, (short)0, c1, false, false);
        c0 = __builtin_amdgcn_wmma_f32_16x16x32_bf16(false, a01, false, bf1, (short)0, c0, false, false);
        c1 = __builtin_amdgcn_wmma_f32_16x16x32_bf16(false, a11, false, bf1, (short)0, c1, false, false);

        const int col = nt * 16 + mcol;
        #pragma unroll
        for (int r = 0; r < 8; ++r) {
            const float v0 = c0[r] * csA[nt] + csB[nt];
            if (v0 > 0.0f)                    // relu(x)<=0 never beats init 0
                atomicMax((unsigned int*)gf + (gbase0[r] + col), __float_as_uint(v0));
            const float v1 = c1[r] * csA[nt] + csB[nt];
            if (v1 > 0.0f)
                atomicMax((unsigned int*)gf + (gbase1[r] + col), __float_as_uint(v1));
        }
    }
}

// ---------------------------------------------------------------------------
// Kernel B: FC1 (32,65536)@(65536,512), split-K WMMA bf16, atomicAdd f32.
// One wave per block. blockIdx.x: N-group (128 cols), blockIdx.y: K split.
// HBM-bound on wf1 (134 MB); A re-read only 4x thanks to 8 N-tiles/block.
// ---------------------------------------------------------------------------
#define KSPLIT 256
#define KPB    (65536 / KSPLIT)   // 256 -> 8 k-steps of 32

__global__ __launch_bounds__(32) void fc1_kernel(
    const float* __restrict__ gf,    // (32, 65536) row-major
    const float* __restrict__ wf1,   // (65536, 512) row-major
    float* __restrict__ acc)         // (32, 512) zero-initialized
{
    const int lane = threadIdx.x;
    const int mcol = lane & 15;
    const int kh   = lane >> 4;
    const int ng   = blockIdx.x;               // 0..3 -> cols [ng*128, +128)
    const int k0b  = blockIdx.y * KPB;

    v8f cacc[2][8];
    #pragma unroll
    for (int t = 0; t < 2; ++t)
        #pragma unroll
        for (int nt = 0; nt < 8; ++nt)
            cacc[t][nt] = (v8f){};

    for (int step = 0; step < KPB / 32; ++step) {
        const int k0 = k0b + step * 32;

        v16bf a[2];                            // two M-tiles of A (rows 0-15, 16-31)
        #pragma unroll
        for (int t = 0; t < 2; ++t) {
            const float* gp = gf + (size_t)(t * 16 + mcol) * 65536 + k0;
            #pragma unroll
            for (int v = 0; v < 8; ++v) {
                const int kb = ((v & 4) ? 16 : 0) + kh * 8 + 2 * (v & 3);
                const float2 f = *(const float2*)(gp + kb);
                a[t][2 * v]     = (__bf16)f.x;
                a[t][2 * v + 1] = (__bf16)f.y;
            }
        }

        #pragma unroll
        for (int nt = 0; nt < 8; ++nt) {
            const int col = ng * 128 + nt * 16 + mcol;
            v16bf bb;
            #pragma unroll
            for (int v = 0; v < 8; ++v) {
                const int kb = ((v & 4) ? 16 : 0) + kh * 8 + 2 * (v & 3);
                bb[2 * v]     = (__bf16)wf1[(size_t)(k0 + kb)     * 512 + col];
                bb[2 * v + 1] = (__bf16)wf1[(size_t)(k0 + kb + 1) * 512 + col];
            }
            cacc[0][nt] = __builtin_amdgcn_wmma_f32_16x16x32_bf16(false, a[0], false, bb, (short)0, cacc[0][nt], false, false);
            cacc[1][nt] = __builtin_amdgcn_wmma_f32_16x16x32_bf16(false, a[1], false, bb, (short)0, cacc[1][nt], false, false);
        }
    }

    #pragma unroll
    for (int t = 0; t < 2; ++t)
        #pragma unroll
        for (int nt = 0; nt < 8; ++nt) {
            const int col = ng * 128 + nt * 16 + mcol;
            #pragma unroll
            for (int r = 0; r < 8; ++r) {
                const int row = t * 16 + r + 8 * kh;
                atomicAdd(&acc[row * 512 + col], cacc[t][nt][r]);
            }
        }
}

// ---------------------------------------------------------------------------
// Kernel C: bn/relu on FC1 output, FC2 (512->256)+bn/relu, FC3 (256->40).
// Single block, LDS-resident intermediates (tiny FLOPs).
// ---------------------------------------------------------------------------
__global__ __launch_bounds__(256) void head_kernel(
    const float* __restrict__ acc,   // (32, 512)
    const float* __restrict__ bf1,
    const float* __restrict__ g3, const float* __restrict__ be3,
    const float* __restrict__ m3, const float* __restrict__ v3,
    const float* __restrict__ wf2,   // (512, 256)
    const float* __restrict__ bf2,
    const float* __restrict__ g4, const float* __restrict__ be4,
    const float* __restrict__ m4, const float* __restrict__ v4,
    const float* __restrict__ wf3,   // (256, 40)
    const float* __restrict__ bf3,
    float* __restrict__ out)         // (32, 40)
{
    __shared__ float h1[32 * 512];
    __shared__ float h2[32 * 256];
    const int tid = threadIdx.x;

    for (int i = tid; i < 32 * 512; i += 256) {
        const int j = i & 511;
        const float s = g3[j] * rsqrtf(v3[j] + EPS);
        h1[i] = fmaxf((acc[i] + bf1[j] - m3[j]) * s + be3[j], 0.0f);
    }
    __syncthreads();

    for (int i = tid; i < 32 * 256; i += 256) {
        const int r = i >> 8, j = i & 255;
        float d = 0.0f;
        for (int k = 0; k < 512; ++k)
            d = fmaf(h1[r * 512 + k], wf2[k * 256 + j], d);
        const float s = g4[j] * rsqrtf(v4[j] + EPS);
        h2[i] = fmaxf((d + bf2[j] - m4[j]) * s + be4[j], 0.0f);
    }
    __syncthreads();

    for (int i = tid; i < 32 * 40; i += 256) {
        const int r = i / 40, j = i % 40;
        float d = bf3[j];
        for (int k = 0; k < 256; ++k)
            d = fmaf(h2[r * 256 + k], wf3[k * 40 + j], d);
        out[i] = d;
    }
}

// ---------------------------------------------------------------------------
extern "C" void kernel_launch(void* const* d_in, const int* in_sizes, int n_in,
                              void* d_out, int out_size, void* d_ws, size_t ws_size,
                              hipStream_t stream) {
    const float* points = (const float*)d_in[0];
    const float* w1  = (const float*)d_in[1];
    const float* b1  = (const float*)d_in[2];
    const float* g1  = (const float*)d_in[3];
    const float* be1 = (const float*)d_in[4];
    const float* m1  = (const float*)d_in[5];
    const float* v1  = (const float*)d_in[6];
    const float* w2  = (const float*)d_in[7];
    const float* b2  = (const float*)d_in[8];
    const float* g2  = (const float*)d_in[9];
    const float* be2 = (const float*)d_in[10];
    const float* m2  = (const float*)d_in[11];
    const float* v2  = (const float*)d_in[12];
    const float* wf1 = (const float*)d_in[13];
    const float* bf1 = (const float*)d_in[14];
    const float* g3  = (const float*)d_in[15];
    const float* be3 = (const float*)d_in[16];
    const float* m3  = (const float*)d_in[17];
    const float* v3  = (const float*)d_in[18];
    const float* wf2 = (const float*)d_in[19];
    const float* bf2 = (const float*)d_in[20];
    const float* g4  = (const float*)d_in[21];
    const float* be4 = (const float*)d_in[22];
    const float* m4  = (const float*)d_in[23];
    const float* v4  = (const float*)d_in[24];
    const float* wf3 = (const float*)d_in[25];
    const float* bf3 = (const float*)d_in[26];

    float* gf  = (float*)d_ws;                 // 32*512*128 = 2,097,152 floats
    float* acc = gf + (BB * GV * 128);         // 32*512     =    16,384 floats
    const int nzero = BB * GV * 128 + BB * 512;

    zero_kernel<<<(nzero + 255) / 256, 256, 0, stream>>>(gf, nzero);

    mlp_pool_kernel<<<(BB * NN) / 256, 256, 0, stream>>>(
        points, w1, b1, g1, be1, m1, v1, w2, b2, g2, be2, m2, v2, gf);

    dim3 fcgrid(4, KSPLIT);
    fc1_kernel<<<fcgrid, 32, 0, stream>>>(gf, wf1, acc);

    head_kernel<<<1, 256, 0, stream>>>(
        acc, bf1, g3, be3, m3, v3, wf2, bf2, g4, be4, m4, v4, wf3, bf3,
        (float*)d_out);
}
